// ScaledDotProductAttention_16724602651288
// MI455X (gfx1250) — matrix operand order repaired and checked by reading, not compiled, not run on hardware
//
#include <hip/hip_runtime.h>
#include <hip/hip_bf16.h>

typedef __attribute__((ext_vector_type(16))) __bf16 v16bf;
typedef __attribute__((ext_vector_type(8)))  float  v8f;

#define B_ 2
#define H_ 16
#define S_ 2048
#define D_ 64
#define MTILE 32           // query rows per workgroup
#define NEGV (-1e9f)

__device__ __forceinline__ float wave_max32(float v) {
#pragma unroll
  for (int off = 16; off > 0; off >>= 1)
    v = fmaxf(v, __shfl_xor(v, off, 32));
  return v;
}
__device__ __forceinline__ float wave_sum32(float v) {
#pragma unroll
  for (int off = 16; off > 0; off >>= 1)
    v += __shfl_xor(v, off, 32);
  return v;
}

// One workgroup = 32 query rows of one (b,h). 8 waves (wave32).
// LDS: 32 x 2048 f32 score stripe = 256 KB (dynamic shared, <= 320 KB/WGP).
__global__ void sdpa_fwd_wmma(const float* __restrict__ q,
                              const float* __restrict__ k,
                              const float* __restrict__ v,
                              const int*   __restrict__ mask,
                              float* __restrict__ out,    // [B,H,S,D]
                              float* __restrict__ attn)   // [B,H,S,S]
{
  extern __shared__ float slds[];   // [MTILE][S_]

  const int lane = threadIdx.x & 31;
  const int wave = threadIdx.x >> 5;     // 0..7
  const int lo   = lane & 15;
  const int hi   = lane >> 4;            // lane group (0: lanes 0-15, 1: 16-31)

  const int tiles_per_bh = S_ / MTILE;   // 64
  const int bh    = blockIdx.x / tiles_per_bh;
  const int qbase = (blockIdx.x % tiles_per_bh) * MTILE;

  // ---------------- Phase 1: scores = Q K^T / sqrt(D) into LDS ----------------
  {
    const int rb = wave >> 2;            // row block (16 rows each)
    const int kg = wave & 3;             // key-tile group (strided over key tiles)

    // Q A-fragments (16x32 bf16, ISA A layout: per lane two runs of 8:
    //   k = hi*8 + [0..7] and k = 16 + hi*8 + [0..7]), one per 32-wide d-chunk.
    v16bf aq[2];
    {
      const float* qp = q + ((size_t)bh * S_ + qbase + rb * 16 + lo) * D_;
#pragma unroll
      for (int c = 0; c < 2; ++c) {
        alignas(16) float t[16];
        const float* p0 = qp + c * 32 + hi * 8;
        *(float4*)&t[0]  = *(const float4*)(p0);
        *(float4*)&t[4]  = *(const float4*)(p0 + 4);
        *(float4*)&t[8]  = *(const float4*)(p0 + 16);
        *(float4*)&t[12] = *(const float4*)(p0 + 20);
#pragma unroll
        for (int i = 0; i < 16; ++i) aq[c][i] = (__bf16)t[i];
      }
    }

    for (int kt = kg; kt < S_ / 16; kt += 4) {
      const int key0 = kt * 16;
      v8f acc = {};
#pragma unroll
      for (int c = 0; c < 2; ++c) {
        // B = K^T chunk (32x16): B(kk,n) = K[key0+n][c*32+kk].
        // Lane: n = lo, kk = hi*8 + [0..7] and 16 + hi*8 + [0..7] (matches A).
        alignas(16) float t[16];
        const float* kp = k + ((size_t)bh * S_ + key0 + lo) * D_ + c * 32 + hi * 8;
        *(float4*)&t[0]  = *(const float4*)(kp);
        *(float4*)&t[4]  = *(const float4*)(kp + 4);
        *(float4*)&t[8]  = *(const float4*)(kp + 16);
        *(float4*)&t[12] = *(const float4*)(kp + 20);
        v16bf bk;
#pragma unroll
        for (int i = 0; i < 16; ++i) bk[i] = (__bf16)t[i];
        acc = __builtin_amdgcn_wmma_f32_16x16x32_bf16(false, aq[c], false, bk,
                                                      (short)0, acc, false, false);
      }
      // C/D layout: VGPR r -> row m = r + 8*hi, col n = lo
#pragma unroll
      for (int r = 0; r < 8; ++r)
        slds[(rb * 16 + r + hi * 8) * S_ + key0 + lo] = acc[r] * 0.125f;
    }
  }
  __syncthreads();

  // ---------------- Phase 2: mask + softmax, write attn ----------------
  {
#pragma unroll
    for (int rr = 0; rr < MTILE / 8; ++rr) {
      const int row  = wave * (MTILE / 8) + rr;       // 0..31
      const int qrow = qbase + row;
      float* srow = slds + row * S_;
      const int* mrow = mask + (size_t)qrow * S_;

      float mx = -3.0e38f;
      for (int c = lane; c < S_; c += 32) {
        float s = mrow[c] ? srow[c] : NEGV;
        srow[c] = s;
        mx = fmaxf(mx, s);
      }
      mx = wave_max32(mx);

      float sum = 0.f;
      for (int c = lane; c < S_; c += 32) {
        float e = __expf(srow[c] - mx);
        srow[c] = e;
        sum += e;
      }
      sum = wave_sum32(sum);
      const float inv = 1.0f / sum;

      float* arow = attn + ((size_t)bh * S_ + qrow) * S_;
      for (int c = lane; c < S_; c += 32) {
        float p = srow[c] * inv;
        srow[c] = p;                          // keep normalized P in LDS for phase 3
        __builtin_nontemporal_store(p, arow + c);  // stream attn (write-once) with TH=NT
      }
    }
  }
  __syncthreads();

  // ---------------- Phase 3: O = P V ----------------
  {
    const int rb   = wave >> 2;            // row block
    const int dcol = (wave & 3) * 16;      // output column tile
    v8f oacc = {};
    for (int kt = 0; kt < S_ / 32; ++kt) {
      const int key0 = kt * 32;
      // A = P chunk (16x32 over keys): lane m = lo; runs at key0+hi*8 and key0+16+hi*8
      alignas(16) float t[16];
      const float* pr = slds + (rb * 16 + lo) * S_ + key0 + hi * 8;
      *(float4*)&t[0]  = *(const float4*)(pr);
      *(float4*)&t[4]  = *(const float4*)(pr + 4);
      *(float4*)&t[8]  = *(const float4*)(pr + 16);
      *(float4*)&t[12] = *(const float4*)(pr + 20);
      v16bf ap;
#pragma unroll
      for (int i = 0; i < 16; ++i) ap[i] = (__bf16)t[i];
      // B = V chunk (32x16): B(kk,n) = V[key0+kk][dcol+n]; lane n = lo, kk = hi*8+i (i<8), 16+hi*8+(i-8)
      const float* vp = v + ((size_t)bh * S_ + key0 + hi * 8) * D_ + dcol + lo;
      v16bf bv;
#pragma unroll
      for (int i = 0; i < 16; ++i) bv[i] = (__bf16)vp[(size_t)(i + (i >= 8 ? 8 : 0)) * D_];
      oacc = __builtin_amdgcn_wmma_f32_16x16x32_bf16(false, ap, false, bv,
                                                     (short)0, oacc, false, false);
    }
    float* op = out + ((size_t)bh * S_ + qbase + rb * 16 + hi * 8) * D_ + dcol + lo;
#pragma unroll
    for (int r = 0; r < 8; ++r)
      __builtin_nontemporal_store(oacc[r], op + (size_t)r * D_);
  }
}

extern "C" void kernel_launch(void* const* d_in, const int* in_sizes, int n_in,
                              void* d_out, int out_size, void* d_ws, size_t ws_size,
                              hipStream_t stream) {
  const float* q    = (const float*)d_in[0];
  const float* k    = (const float*)d_in[1];
  const float* v    = (const float*)d_in[2];
  const int*   mask = (const int*)d_in[3];

  float* out  = (float*)d_out;                       // [B,H,S,D] first
  float* attn = out + (size_t)B_ * H_ * S_ * D_;     // then [B,H,S,S]

  const size_t shmem = (size_t)MTILE * S_ * sizeof(float);  // 256 KB (<= 320 KB/WGP)
  hipFuncSetAttribute(reinterpret_cast<const void*>(sdpa_fwd_wmma),
                      hipFuncAttributeMaxDynamicSharedMemorySize, (int)shmem);

  dim3 grid(B_ * H_ * (S_ / MTILE));   // 2048 workgroups
  dim3 block(256);                     // 8 wave32 waves
  sdpa_fwd_wmma<<<grid, block, shmem, stream>>>(q, k, v, mask, out, attn);
}
